// WindowAttention_16733192585299
// MI455X (gfx1250) — matrix operand order repaired and checked by reading, not compiled, not run on hardware
//
#include <hip/hip_runtime.h>
#include <hip/hip_bf16.h>

#define WIN 7
#define S 49
#define SP 64
#define HEADS 4
#define HD 32
#define C 128
#define NQKV 384
#define NWIN 64
#define SCALE 0.17677669529663687f

typedef __attribute__((ext_vector_type(16))) _Float16 v16h;
typedef __attribute__((ext_vector_type(8)))  _Float16 v8h;
typedef __attribute__((ext_vector_type(8)))  float    v8f;
typedef __attribute__((ext_vector_type(4)))  unsigned int v4u;
typedef __attribute__((ext_vector_type(8)))  int v8i;
typedef __attribute__((ext_vector_type(4)))  int v4i;

// ---- Fragment layout helpers (contiguous k order used by this code, wave32) ----
// A-frag 16x32 f16: lane L (m = L&15, hi = L>>4); element e: k = (hi<<4)+e
__device__ __forceinline__ int afrag_lane(int mm, int kk) { return mm + ((kk >> 4) << 4); }
__device__ __forceinline__ int afrag_elem(int kk)         { return kk & 15; }
// B-frag 32x16 f16: lane L (n = L&15, kbase = (L>>4)*16); element e: k = kbase + e
__device__ __forceinline__ int bfrag_lane(int kk, int nn) { return nn + ((kk >> 4) << 4); }
__device__ __forceinline__ int bfrag_elem(int kk)         { return kk & 15; }
// C/D f32 frag 16x16: lane L, reg r -> m = r + 8*(L>>4), n = L&15

// ---------------- prep: weights -> f16 fragment layout, bias gather ----------------
__global__ __launch_bounds__(256) void wa_prep_kernel(
    const float* __restrict__ w_qkv, const float* __restrict__ w_proj,
    const int* __restrict__ rel_index, const float* __restrict__ bias_table,
    _Float16* __restrict__ wqkv_f, _Float16* __restrict__ wproj_f,
    float* __restrict__ bias_f)
{
  int tid = blockIdx.x * blockDim.x + threadIdx.x;
  int total = C * NQKV + C * C + HEADS * S * S;
  for (int i = tid; i < total; i += gridDim.x * blockDim.x) {
    if (i < C * NQKV) {
      int k = i / NQKV, n = i % NQKV;
      int kc = k >> 5, kk = k & 31, nt = n >> 4, nn = n & 15;
      wqkv_f[(kc * 24 + nt) * 512 + bfrag_lane(kk, nn) * 16 + bfrag_elem(kk)] =
          (_Float16)w_qkv[i];
    } else if (i < C * NQKV + C * C) {
      int j = i - C * NQKV;
      int k = j / C, n = j % C;
      int kc = k >> 5, kk = k & 31, nt = n >> 4, nn = n & 15;
      wproj_f[(kc * 8 + nt) * 512 + bfrag_lane(kk, nn) * 16 + bfrag_elem(kk)] =
          (_Float16)w_proj[j];
    } else {
      int j = i - C * NQKV - C * C;          // [h][49*49]
      int h = j / (S * S);
      int rc = j % (S * S);
      bias_f[j] = bias_table[rel_index[rc] * HEADS + h];
    }
  }
}

// ---------------- fused window attention ----------------
// LDS carve (halves): xA 8192 | qA 8192 | kB 8192 | vB 8192 | attnA 16384 | outA 8192
// then attnF (f32, 4*64*65) then xStage (f32, 49*128 TDM landing pad)
#define LDS_ATTNF_HALVES 57344
#define LDS_XSTAGE_HALVES 90624          // 57344 + 4*64*65*2
#define LDS_TOTAL_BYTES (181248 + 25088) // 206336

__global__ __launch_bounds__(256) void wa_main_kernel(
    const float* __restrict__ x, const float* __restrict__ mask,
    const float* __restrict__ b_qkv, const float* __restrict__ b_proj,
    const _Float16* __restrict__ wqkv_f, const _Float16* __restrict__ wproj_f,
    const float* __restrict__ bias_f, float* __restrict__ out)
{
  extern __shared__ _Float16 smem[];
  _Float16* xA    = smem;
  _Float16* qA    = smem + 8192;
  _Float16* kB    = smem + 16384;
  _Float16* vB    = smem + 24576;
  _Float16* attnA = smem + 32768;
  _Float16* outA  = smem + 49152;
  float*    attnF = (float*)(smem + LDS_ATTNF_HALVES);   // stride 65 vs bank conflicts
  float*    xStage= (float*)(smem + LDS_XSTAGE_HALVES);

  const int blk = blockIdx.x;
  const int tid = threadIdx.x;
  const int wv = tid >> 5;
  const int ln = tid & 31;

  // ---- Stage 0a: TDM async DMA of the x window (49x128 f32) into LDS ----
  const float* xw = x + (size_t)blk * S * C;
#if __has_builtin(__builtin_amdgcn_tensor_load_to_lds)
  if (wv == 0) {
    unsigned long long ga = (unsigned long long)(uintptr_t)xw;
    unsigned int lds_addr = (unsigned int)(uintptr_t)(void*)xStage;
    v4u g0;
    g0[0] = 1u;                                        // count=1, user mode
    g0[1] = lds_addr;                                  // lds_addr[31:0]
    g0[2] = (unsigned int)(ga & 0xFFFFFFFFull);        // global_addr[31:0]
    g0[3] = (unsigned int)((ga >> 32) & 0x01FFFFFFull) // global_addr[56:32]
            | 0x80000000u;                             // type=2 ("image")
    v8i g1;
    g1[0] = 0x00020000;            // workgroup_mask=0, data_size=2 (4B)
    g1[1] = (int)(128u << 16);     // tensor_dim0=128 (low16 @ [31:16])
    g1[2] = (int)(49u << 16);      // tensor_dim0 hi=0, tensor_dim1=49 (low16 @ [31:16])
    g1[3] = (int)(128u << 16);     // tensor_dim1 hi=0, tile_dim0=128
    g1[4] = 49;                    // tile_dim1=49, tile_dim2=0
    g1[5] = 128;                   // tensor_dim0_stride[31:0]=128
    g1[6] = 0;                     // stride0 hi, stride1 lo
    g1[7] = 0;
    v4i g2 = {0, 0, 0, 0}, g3 = {0, 0, 0, 0};
    v8i g4 = {0, 0, 0, 0, 0, 0, 0, 0};
    __builtin_amdgcn_tensor_load_to_lds(g0, g1, g2, g3, g4, 0);
    __builtin_amdgcn_s_wait_tensorcnt(0);
  }
#else
  if (wv == 0 && ln == 0) {
    for (int i = 0; i < S * C; ++i) xStage[i] = xw[i];  // never taken on gfx1250
  }
#endif
  if (tid < 96) __builtin_prefetch(wqkv_f + tid * 512, 0, 3);   // global_prefetch_b8
  __syncthreads();

  // ---- Stage 0b: convert staged x to f16 A-fragments (64x128, zero-pad rows) ----
  for (int g = tid; g < (SP * C) / 8; g += 256) {
    int m  = g >> 4;           // row
    int k0 = (g & 15) << 3;    // col group of 8
    v8h hv;
    if (m < S) {
      const float4* p4 = (const float4*)(xStage + m * C + k0);
      float4 u0 = p4[0], u1 = p4[1];
      hv[0] = (_Float16)u0.x; hv[1] = (_Float16)u0.y;
      hv[2] = (_Float16)u0.z; hv[3] = (_Float16)u0.w;
      hv[4] = (_Float16)u1.x; hv[5] = (_Float16)u1.y;
      hv[6] = (_Float16)u1.z; hv[7] = (_Float16)u1.w;
    } else {
      hv = (v8h)(_Float16)0.0f;
    }
    int mt = m >> 4, kc = k0 >> 5, mm = m & 15, kk = k0 & 31;
    *(v8h*)(xA + (mt * 4 + kc) * 512 + afrag_lane(mm, kk) * 16 + afrag_elem(kk)) = hv;
  }
  __syncthreads();

  // ---- Stage 1: QKV = x @ w_qkv + b_qkv; scatter into q/k/v fragments ----
  for (int t = 0; t < 12; ++t) {
    int tile = wv * 12 + t;           // 0..95
    int mt = tile / 24, nt = tile % 24;
    v8f acc = {};
#pragma unroll
    for (int kc = 0; kc < 4; ++kc) {
      v16h a = *(const v16h*)(xA + (mt * 4 + kc) * 512 + ln * 16);
      v16h b = *(const v16h*)(wqkv_f + (kc * 24 + nt) * 512 + ln * 16);
      acc = __builtin_amdgcn_wmma_f32_16x16x32_f16(false, a, false, b, (short)0, acc,
                                                   false, false);
    }
    // tile-uniform routing: sel/h/dbase depend only on nt
    int nn = ln & 15;
    int hi = ln >> 4;
    float bq = b_qkv[nt * 16 + nn];
    int sel = nt >> 3;                 // 0=q 1=k 2=v
    int h = (nt >> 1) & 3;
    int d = ((nt & 1) << 4) + nn;      // 0..31
    if (sel == 0) {                    // q -> A-frag (per head, K = d)
      _Float16* base = qA + (h * 4 + mt) * 512;
      int lp = ((d >> 4) << 4);
      int e = afrag_elem(d);
#pragma unroll
      for (int r = 0; r < 8; ++r) {
        int mm = (r + (hi << 3)) & 15;
        base[(mm + lp) * 16 + e] = (_Float16)(acc[r] + bq);
      }
    } else if (sel == 1) {             // k -> B-frag of k^T (kdim=d, col=token)
      _Float16* base = kB + (h * 4 + mt) * 512;
      int lp = ((d >> 4) << 4);
      int e = d & 15;
#pragma unroll
      for (int r = 0; r < 8; ++r) {
        int mm = (r + (hi << 3)) & 15;
        base[(mm + lp) * 16 + e] = (_Float16)(acc[r] + bq);
      }
    } else {                           // v -> B-frag (kdim=token, col=d)
      int nt2 = d >> 4, nn2 = d & 15;
#pragma unroll
      for (int r = 0; r < 8; ++r) {
        int m = mt * 16 + r + (hi << 3);
        int kc2 = m >> 5, kk2 = m & 31;
        vB[(h * 4 + kc2 * 2 + nt2) * 512 + bfrag_lane(kk2, nn2) * 16 + (kk2 & 15)] =
            (_Float16)(acc[r] + bq);
      }
    }
  }
  __syncthreads();

  // ---- Stage 2: attn = scale * q k^T + bias + mask -> attnF (f32) ----
  const float* maskw = mask + (size_t)(blk & (NWIN - 1)) * S * S;
  for (int t = 0; t < 8; ++t) {
    int tile = wv * 8 + t;            // 0..63
    int h = tile >> 4, r2 = tile & 15, mt = r2 >> 2, nt = r2 & 3;
    v16h a = *(const v16h*)(qA + (h * 4 + mt) * 512 + ln * 16);
    v16h b = *(const v16h*)(kB + (h * 4 + nt) * 512 + ln * 16);
    v8f acc = {};
    acc = __builtin_amdgcn_wmma_f32_16x16x32_f16(false, a, false, b, (short)0, acc,
                                                 false, false);
    int n = nt * 16 + (ln & 15);
#pragma unroll
    for (int r = 0; r < 8; ++r) {
      int m = mt * 16 + r + ((ln >> 4) << 3);
      float val = -1e30f;
      if (m < S && n < S)
        val = acc[r] * SCALE + bias_f[h * S * S + m * S + n] + maskw[m * S + n];
      attnF[(h * SP + m) * 65 + n] = val;
    }
  }
  __syncthreads();

  // ---- Stage 3: softmax per (head,row), write f16 A-fragments of probs ----
  {
    int h = tid >> 6;                 // 0..3
    int row = tid & 63;
    float* rp = attnF + (h * SP + row) * 65;
    int mt = row >> 4, mm = row & 15;
    if (row < S) {
      float mx = -1e30f;
      for (int j = 0; j < S; ++j) mx = fmaxf(mx, rp[j]);
      float sum = 0.f;
      for (int j = 0; j < S; ++j) { float e = __expf(rp[j] - mx); rp[j] = e; sum += e; }
      float rinv = 1.f / sum;
#pragma unroll
      for (int g = 0; g < 8; ++g) {
        v8h hv;
#pragma unroll
        for (int j = 0; j < 8; ++j) {
          int col = g * 8 + j;
          hv[j] = (_Float16)((col < S) ? rp[col] * rinv : 0.f);
        }
        int kk = (g * 8) & 31, kc = g >> 2;
        *(v8h*)(attnA + ((h * 4 + mt) * 2 + kc) * 512 + afrag_lane(mm, kk) * 16 +
                afrag_elem(kk)) = hv;
      }
    } else {
      v8h z = (v8h)(_Float16)0.0f;
#pragma unroll
      for (int g = 0; g < 8; ++g) {
        int kk = (g * 8) & 31, kc = g >> 2;
        *(v8h*)(attnA + ((h * 4 + mt) * 2 + kc) * 512 + afrag_lane(mm, kk) * 16 +
                afrag_elem(kk)) = z;
      }
    }
  }
  __syncthreads();

  // ---- Stage 4: out_h = probs @ v ; scatter [64 x 128] as f16 A-fragments ----
  for (int t = 0; t < 4; ++t) {
    int tile = wv * 4 + t;            // 0..31
    int h = tile >> 3, r3 = tile & 7, mt = r3 >> 1, ntd = r3 & 1;
    v8f acc = {};
#pragma unroll
    for (int kc = 0; kc < 2; ++kc) {
      v16h a = *(const v16h*)(attnA + ((h * 4 + mt) * 2 + kc) * 512 + ln * 16);
      v16h b = *(const v16h*)(vB + (h * 4 + kc * 2 + ntd) * 512 + ln * 16);
      acc = __builtin_amdgcn_wmma_f32_16x16x32_f16(false, a, false, b, (short)0, acc,
                                                   false, false);
    }
    int dcol = h * 32 + ntd * 16 + (ln & 15);
    int kc2 = dcol >> 5, kkd = dcol & 31;
    int lp = ((kkd >> 4) << 4);
    int e = afrag_elem(kkd);
#pragma unroll
    for (int r = 0; r < 8; ++r) {
      int mm = (r + ((ln >> 4) << 3)) & 15;
      outA[(mt * 4 + kc2) * 512 + (mm + lp) * 16 + e] = (_Float16)acc[r];
    }
  }
  __syncthreads();

  // ---- Stage 5: y = out @ w_proj + b_proj -> global ----
  float* ow = out + (size_t)blk * S * C;
  for (int t = 0; t < 4; ++t) {
    int tile = wv * 4 + t;            // 0..31
    int mt = tile >> 3, nt = tile & 7;
    v8f acc = {};
#pragma unroll
    for (int kc = 0; kc < 4; ++kc) {
      v16h a = *(const v16h*)(outA + (mt * 4 + kc) * 512 + ln * 16);
      v16h b = *(const v16h*)(wproj_f + (kc * 8 + nt) * 512 + ln * 16);
      acc = __builtin_amdgcn_wmma_f32_16x16x32_f16(false, a, false, b, (short)0, acc,
                                                   false, false);
    }
    int n = nt * 16 + (ln & 15);
    float bp = b_proj[n];
#pragma unroll
    for (int r = 0; r < 8; ++r) {
      int m = mt * 16 + r + ((ln >> 4) << 3);
      if (m < S) ow[m * C + n] = acc[r] + bp;
    }
  }
}

extern "C" void kernel_launch(void* const* d_in, const int* in_sizes, int n_in,
                              void* d_out, int out_size, void* d_ws, size_t ws_size,
                              hipStream_t stream) {
  const float* x          = (const float*)d_in[0];
  const float* mask       = (const float*)d_in[1];
  const int*   rel_index  = (const int*)d_in[2];
  const float* bias_table = (const float*)d_in[3];
  const float* w_qkv      = (const float*)d_in[4];
  const float* b_qkv      = (const float*)d_in[5];
  const float* w_proj     = (const float*)d_in[6];
  const float* b_proj     = (const float*)d_in[7];
  float* out = (float*)d_out;

  _Float16* wqkv_f  = (_Float16*)d_ws;                              // 96 tiles * 1KB
  _Float16* wproj_f = (_Float16*)((char*)d_ws + 98304);             // 32 tiles * 1KB
  float*    bias_f  = (float*)((char*)d_ws + 98304 + 32768);        // 4*49*49 f32

  wa_prep_kernel<<<64, 256, 0, stream>>>(w_qkv, w_proj, rel_index, bias_table,
                                         wqkv_f, wproj_f, bias_f);

  wa_main_kernel<<<4096, 256, LDS_TOTAL_BYTES, stream>>>(x, mask, b_qkv, b_proj,
                                                         wqkv_f, wproj_f, bias_f, out);
}
